// Attention_80539226734697
// MI455X (gfx1250) — matrix operand order repaired and checked by reading, not compiled, not run on hardware
//
#include <hip/hip_runtime.h>
#include <hip/hip_bf16.h>

// ---------------------------------------------------------------------------
// MHA forward for MI455X (gfx1250). All GEMMs on v_wmma_f32_16x16x32_bf16
// (f32 accumulate). K-loops use ping-pong (unroll-2) software pipelining with
// two named fragment sets so loads write registers directly (no rotation
// moves, fewer WMMA hazard nops) while loads overlap WMMAs via loadcnt.
// 32x64 per-wave GEMM tiles, LDS softmax, vis_attn via global f32 atomics.
// ---------------------------------------------------------------------------

typedef __bf16 bf16;
typedef __attribute__((ext_vector_type(8)))  bf16  v8bf;
typedef __attribute__((ext_vector_type(16))) bf16  v16bf;
typedef __attribute__((ext_vector_type(8)))  float v8f;

#define DIM    768
#define NHEADS 12
#define HD     64
#define BATCH  8
#define SEQ    1024
#define MTOT   (BATCH * SEQ)       // 8192 rows
#define QKVN   (3 * DIM)           // 2304
#define ATT_SCALE 0.125f           // 1/sqrt(64)
#define INV_BH (1.0f / 96.0f)

// ---- WMMA fragment loaders (contiguous k order: lane half h holds k = 16h..16h+15) -------

__device__ __forceinline__ v16bf load_a_frag(const bf16* base, int ld, int lane) {
  const int row = lane & 15;
  const int kg  = (lane >> 4) << 4;
  const bf16* p0 = base + (size_t)row * ld + kg;
  v8bf lo = *(const v8bf*)p0;
  v8bf hi = *(const v8bf*)(p0 + 8);
  v16bf f;
#pragma unroll
  for (int i = 0; i < 8; ++i) { f[i] = lo[i]; f[i + 8] = hi[i]; }
  return f;
}

__device__ __forceinline__ v16bf load_b_frag(const bf16* base, int ld, int lane) {
  const int col = lane & 15;
  const int kg  = (lane >> 4) << 4;
  const bf16* p0 = base + (size_t)col * ld + kg;
  v8bf lo = *(const v8bf*)p0;
  v8bf hi = *(const v8bf*)(p0 + 8);
  v16bf f;
#pragma unroll
  for (int i = 0; i < 8; ++i) { f[i] = lo[i]; f[i + 8] = hi[i]; }
  return f;
}

__device__ __forceinline__ v8f wmma_bf16(v16bf a, v16bf b, v8f c) {
  return __builtin_amdgcn_wmma_f32_16x16x32_bf16(false, a, false, b,
                                                 (short)0, c, false, false);
}

// ---- prep kernels ----------------------------------------------------------

__global__ __launch_bounds__(256) void cvt_f32_bf16_kernel(
    const float* __restrict__ src, bf16* __restrict__ dst, int n) {
  for (int i = blockIdx.x * 256 + threadIdx.x; i < n; i += gridDim.x * 256)
    dst[i] = (bf16)src[i];
}

__global__ __launch_bounds__(256) void zero_f32_kernel(float* __restrict__ dst, int n) {
  for (int i = blockIdx.x * 256 + threadIdx.x; i < n; i += gridDim.x * 256)
    dst[i] = 0.0f;
}

// ---- ping-pong pipelined GEMM body: each wave computes a 32x64 strip -------
// Even/odd fragment sets; loads for the next phase are issued before the
// current phase's 8 WMMAs. No fragment rotation -> no register moves.

#define GEMM_STEP(AF0, AF1, B0, B1, B2, B3)                                    \
  accL[0] = wmma_bf16(AF0, B0, accL[0]);                                       \
  accH[0] = wmma_bf16(AF1, B0, accH[0]);                                       \
  accL[1] = wmma_bf16(AF0, B1, accL[1]);                                       \
  accH[1] = wmma_bf16(AF1, B1, accH[1]);                                       \
  accL[2] = wmma_bf16(AF0, B2, accL[2]);                                       \
  accH[2] = wmma_bf16(AF1, B2, accH[2]);                                       \
  accL[3] = wmma_bf16(AF0, B3, accL[3]);                                       \
  accH[3] = wmma_bf16(AF1, B3, accH[3]);

#define GEMM_BODY(A0PTR, A1PTR, BPTR)                                          \
  v8f accL[4] = {{}, {}, {}, {}};                                              \
  v8f accH[4] = {{}, {}, {}, {}};                                              \
  v16bf ea0 = load_a_frag(A0PTR, DIM, lane);                                   \
  v16bf ea1 = load_a_frag(A1PTR, DIM, lane);                                   \
  v16bf eb0 = load_b_frag(BPTR + 0 * 16 * DIM, DIM, lane);                     \
  v16bf eb1 = load_b_frag(BPTR + 1 * 16 * DIM, DIM, lane);                     \
  v16bf eb2 = load_b_frag(BPTR + 2 * 16 * DIM, DIM, lane);                     \
  v16bf eb3 = load_b_frag(BPTR + 3 * 16 * DIM, DIM, lane);                     \
  for (int k = 0; k < DIM; k += 64) {                                          \
    __builtin_prefetch(A0PTR + k + 64, 0, 3);                                  \
    v16bf oa0 = load_a_frag(A0PTR + k + 32, DIM, lane);                        \
    v16bf oa1 = load_a_frag(A1PTR + k + 32, DIM, lane);                        \
    v16bf ob0 = load_b_frag(BPTR + 0 * 16 * DIM + k + 32, DIM, lane);          \
    v16bf ob1 = load_b_frag(BPTR + 1 * 16 * DIM + k + 32, DIM, lane);          \
    v16bf ob2 = load_b_frag(BPTR + 2 * 16 * DIM + k + 32, DIM, lane);          \
    v16bf ob3 = load_b_frag(BPTR + 3 * 16 * DIM + k + 32, DIM, lane);          \
    GEMM_STEP(ea0, ea1, eb0, eb1, eb2, eb3)                                    \
    if (k + 64 < DIM) {                                                        \
      ea0 = load_a_frag(A0PTR + k + 64, DIM, lane);                            \
      ea1 = load_a_frag(A1PTR + k + 64, DIM, lane);                            \
      eb0 = load_b_frag(BPTR + 0 * 16 * DIM + k + 64, DIM, lane);              \
      eb1 = load_b_frag(BPTR + 1 * 16 * DIM + k + 64, DIM, lane);              \
      eb2 = load_b_frag(BPTR + 2 * 16 * DIM + k + 64, DIM, lane);              \
      eb3 = load_b_frag(BPTR + 3 * 16 * DIM + k + 64, DIM, lane);              \
    }                                                                          \
    GEMM_STEP(oa0, oa1, ob0, ob1, ob2, ob3)                                    \
  }

// ---- QKV projection: qkv = x @ Wqkv^T + b, scattered to q / k / v^T --------
__global__ __launch_bounds__(256) void gemm_qkv_kernel(
    const bf16* __restrict__ xbf, const bf16* __restrict__ wbf,
    const float* __restrict__ bias, bf16* __restrict__ qb,
    bf16* __restrict__ kb, bf16* __restrict__ vtb) {
  const int lane = threadIdx.x & 31;
  const int wave = threadIdx.x >> 5;
  const int row0 = blockIdx.x * 256 + wave * 32;
  const int col0 = blockIdx.y * 64;
  const bf16* aL = xbf + (size_t)row0 * DIM;
  const bf16* aH = aL + (size_t)16 * DIM;
  const bf16* wB = wbf + (size_t)col0 * DIM;
  GEMM_BODY(aL, aH, wB)
  const int mten = (lane >> 4) * 8;
  const int ncol = lane & 15;
#pragma unroll
  for (int half = 0; half < 2; ++half) {
#pragma unroll
    for (int t = 0; t < 4; ++t) {
#pragma unroll
      for (int r = 0; r < 8; ++r) {
        const int i = row0 + half * 16 + mten + r;   // [0, 8192)
        const int j = col0 + t * 16 + ncol;          // [0, 2304)
        const float v = (half ? accH[t][r] : accL[t][r]) + bias[j];
        const int bidx = i >> 10, nn = i & 1023;
        if (j < DIM) {                        // Q: [B,H,N,hd]
          const int h = j >> 6, d = j & 63;
          qb[(((size_t)(bidx * NHEADS + h) << 10) + nn) * HD + d] = (bf16)v;
        } else if (j < 2 * DIM) {             // K: [B,H,N,hd]
          const int jj = j - DIM, h = jj >> 6, d = jj & 63;
          kb[(((size_t)(bidx * NHEADS + h) << 10) + nn) * HD + d] = (bf16)v;
        } else {                              // V transposed: [B,H,hd,N]
          const int jj = j - 2 * DIM, h = jj >> 6, d = jj & 63;
          vtb[((size_t)(bidx * NHEADS + h) * HD + d) * SEQ + nn] = (bf16)v;
        }
      }
    }
  }
}

// ---- attention -------------------------------------------------------------
// grid (64 row-blocks, 96 bh). Dyn LDS: S[16][1024] f32 (64KB) + P bf16 (32KB).
__global__ __launch_bounds__(256) void attn_kernel(
    const bf16* __restrict__ qb, const bf16* __restrict__ kb,
    const bf16* __restrict__ vtb, bf16* __restrict__ aob,
    float* __restrict__ vis) {
  extern __shared__ char smem[];
  float* S = (float*)smem;                       // 16 * 1024 f32
  bf16*  P = (bf16*)(smem + 16 * SEQ * 4);       // 16 * 1024 bf16
  const int rb = blockIdx.x;
  const int bh = blockIdx.y;
  const int b = bh / NHEADS, h = bh % NHEADS;
  const int tid = threadIdx.x;
  const int lane = tid & 31, wave = tid >> 5;
  const bf16* qbh = qb  + (size_t)bh * SEQ * HD;
  const bf16* kbh = kb  + (size_t)bh * SEQ * HD;
  const bf16* vbh = vtb + (size_t)bh * HD * SEQ;
  const int mten = (lane >> 4) * 8;
  const int ncol = lane & 15;

  // Phase 1: S tile (16 x 1024). 8 n-tiles per wave, ping-pong pipelined.
  {
    const bf16* qrow = qbh + (size_t)(rb * 16) * HD;
    v16bf a0 = load_a_frag(qrow + 0,  HD, lane);
    v16bf a1 = load_a_frag(qrow + 32, HD, lane);
    v16bf eb0 = load_b_frag(kbh + (size_t)(wave * 16) * HD + 0,  HD, lane);
    v16bf eb1 = load_b_frag(kbh + (size_t)(wave * 16) * HD + 32, HD, lane);
#pragma unroll
    for (int it = 0; it < 8; it += 2) {
      const int nt0 = wave + it * 8;
      const int nt1 = nt0 + 8;
      const bf16* kt1 = kbh + (size_t)(nt1 * 16) * HD;
      v16bf ob0 = load_b_frag(kt1 + 0,  HD, lane);
      v16bf ob1 = load_b_frag(kt1 + 32, HD, lane);
      v8f c0 = {}, c1 = {};
      c0 = wmma_bf16(a0, eb0, c0);
      c1 = wmma_bf16(a1, eb1, c1);
#pragma unroll
      for (int r = 0; r < 8; ++r)
        S[(size_t)(mten + r) * SEQ + nt0 * 16 + ncol] = (c0[r] + c1[r]) * ATT_SCALE;
      if (it + 2 < 8) {
        const bf16* kt2 = kbh + (size_t)((nt0 + 16) * 16) * HD;
        eb0 = load_b_frag(kt2 + 0,  HD, lane);
        eb1 = load_b_frag(kt2 + 32, HD, lane);
      }
      v8f d0 = {}, d1 = {};
      d0 = wmma_bf16(a0, ob0, d0);
      d1 = wmma_bf16(a1, ob1, d1);
#pragma unroll
      for (int r = 0; r < 8; ++r)
        S[(size_t)(mten + r) * SEQ + nt1 * 16 + ncol] = (d0[r] + d1[r]) * ATT_SCALE;
    }
  }
  __syncthreads();

  // Phase 2: row softmax (16 lanes per row), P to LDS, p/96 -> vis atomics.
  {
    const int row = tid >> 4, sub = tid & 15;
    float* Sr = S + (size_t)row * SEQ;
    float mx = -3.0e38f;
    for (int c = sub; c < SEQ; c += 16) mx = fmaxf(mx, Sr[c]);
#pragma unroll
    for (int o = 8; o >= 1; o >>= 1) mx = fmaxf(mx, __shfl_xor(mx, o, 32));
    float sum = 0.0f;
    for (int c = sub; c < SEQ; c += 16) {
      float e = __expf(Sr[c] - mx);
      Sr[c] = e;
      sum += e;
    }
#pragma unroll
    for (int o = 8; o >= 1; o >>= 1) sum += __shfl_xor(sum, o, 32);
    const float inv = 1.0f / sum;
    float* visrow = vis + (size_t)(rb * 16 + row) * SEQ;
    bf16* Pr = P + (size_t)row * SEQ;
    for (int c = sub; c < SEQ; c += 16) {
      float p = Sr[c] * inv;
      Pr[c] = (bf16)p;
      atomicAdd(&visrow[c], p * INV_BH);
    }
  }
  __syncthreads();

  // Phase 3: out[16x64] = P[16x1024] @ V. 8 waves = (khalf, dtile); ping-pong
  // pipelined over K/2 with even/odd accumulators; partials combined in LDS.
  {
    const int dt = wave & 3;
    const int khalf = wave >> 2;
    const int kbeg = khalf * (SEQ / 2);
    const int kend = kbeg + SEQ / 2;
    const bf16* vrow = vbh + (size_t)(dt * 16) * SEQ;
    v8f c0 = {}, c1 = {};
    v16bf ea = load_a_frag(P + kbeg, SEQ, lane);
    v16bf eb = load_b_frag(vrow + kbeg, SEQ, lane);
    for (int k = kbeg; k < kend; k += 64) {
      v16bf oa = load_a_frag(P + k + 32, SEQ, lane);
      v16bf ob = load_b_frag(vrow + k + 32, SEQ, lane);
      c0 = wmma_bf16(ea, eb, c0);
      if (k + 64 < kend) {
        ea = load_a_frag(P + k + 64, SEQ, lane);
        eb = load_b_frag(vrow + k + 64, SEQ, lane);
      }
      c1 = wmma_bf16(oa, ob, c1);
    }
    float* part = S;          // 8 waves * 32 lanes * 8 f32 = 8KB, S is free
#pragma unroll
    for (int r = 0; r < 8; ++r)
      part[((size_t)wave * 32 + lane) * 8 + r] = c0[r] + c1[r];
    __syncthreads();
    if (wave < 4) {
#pragma unroll
      for (int r = 0; r < 8; ++r) {
        const float v = part[((size_t)wave * 32 + lane) * 8 + r] +
                        part[((size_t)(wave + 4) * 32 + lane) * 8 + r];
        const int i = b * SEQ + rb * 16 + mten + r;
        const int j = h * HD + dt * 16 + ncol;
        aob[(size_t)i * DIM + j] = (bf16)v;
      }
    }
  }
}

// ---- output projection: out = attnout @ Wproj^T + b (fp32 store) ----------
__global__ __launch_bounds__(256) void gemm_proj_kernel(
    const bf16* __restrict__ aob, const bf16* __restrict__ wbf,
    const float* __restrict__ bias, float* __restrict__ out) {
  const int lane = threadIdx.x & 31;
  const int wave = threadIdx.x >> 5;
  const int row0 = blockIdx.x * 256 + wave * 32;
  const int col0 = blockIdx.y * 64;
  const bf16* aL = aob + (size_t)row0 * DIM;
  const bf16* aH = aL + (size_t)16 * DIM;
  const bf16* wB = wbf + (size_t)col0 * DIM;
  GEMM_BODY(aL, aH, wB)
  const int mten = (lane >> 4) * 8;
  const int ncol = lane & 15;
#pragma unroll
  for (int half = 0; half < 2; ++half) {
#pragma unroll
    for (int t = 0; t < 4; ++t) {
#pragma unroll
      for (int r = 0; r < 8; ++r) {
        const int i = row0 + half * 16 + mten + r;
        const int j = col0 + t * 16 + ncol;
        out[(size_t)i * DIM + j] = (half ? accH[t][r] : accL[t][r]) + bias[j];
      }
    }
  }
}

// ---------------------------------------------------------------------------

extern "C" void kernel_launch(void* const* d_in, const int* in_sizes, int n_in,
                              void* d_out, int out_size, void* d_ws, size_t ws_size,
                              hipStream_t stream) {
  const float* x     = (const float*)d_in[0];
  const float* Wqkv  = (const float*)d_in[1];
  const float* bqkv  = (const float*)d_in[2];
  const float* Wproj = (const float*)d_in[3];
  const float* bproj = (const float*)d_in[4];
  float* out = (float*)d_out;
  float* vis = out + (size_t)MTOT * DIM;   // second tuple output [1024,1024]

  char* ws = (char*)d_ws;
  size_t off = 0;
  auto alloc = [&](size_t bytes) -> char* {
    char* p = ws + off;
    off += (bytes + 255) & ~(size_t)255;
    return p;
  };
  bf16* xbf  = (bf16*)alloc((size_t)MTOT * DIM * 2);
  bf16* wqk  = (bf16*)alloc((size_t)QKVN * DIM * 2);
  bf16* wpr  = (bf16*)alloc((size_t)DIM * DIM * 2);
  bf16* qb   = (bf16*)alloc((size_t)MTOT * DIM * 2);   // [B,H,N,hd]
  bf16* kb   = (bf16*)alloc((size_t)MTOT * DIM * 2);   // [B,H,N,hd]
  bf16* vtb  = (bf16*)alloc((size_t)MTOT * DIM * 2);   // [B,H,hd,N]
  bf16* aob  = (bf16*)alloc((size_t)MTOT * DIM * 2);   // [B*N, DIM]

  cvt_f32_bf16_kernel<<<2048, 256, 0, stream>>>(x, xbf, MTOT * DIM);
  cvt_f32_bf16_kernel<<<2048, 256, 0, stream>>>(Wqkv, wqk, QKVN * DIM);
  cvt_f32_bf16_kernel<<<1024, 256, 0, stream>>>(Wproj, wpr, DIM * DIM);
  zero_f32_kernel<<<1024, 256, 0, stream>>>(vis, SEQ * SEQ);

  gemm_qkv_kernel<<<dim3(MTOT / 256, QKVN / 64), 256, 0, stream>>>(
      xbf, wqk, bqkv, qb, kb, vtb);

  const size_t attn_lds = (size_t)16 * SEQ * 4 + (size_t)16 * SEQ * 2;  // 96 KB
  attn_kernel<<<dim3(SEQ / 16, BATCH * NHEADS), 256, attn_lds, stream>>>(
      qb, kb, vtb, aob, vis);

  gemm_proj_kernel<<<dim3(MTOT / 256, DIM / 64), 256, 0, stream>>>(
      aob, wpr, bproj, out);
}
